// TransformerBlock_10084583211371
// MI455X (gfx1250) — matrix operand order repaired and checked by reading, not compiled, not run on hardware
//
#include <hip/hip_runtime.h>
#include <hip/hip_bf16.h>

typedef _Float16 half_t;
typedef __attribute__((ext_vector_type(8)))  _Float16 v8h;
typedef __attribute__((ext_vector_type(16))) _Float16 v16h;
typedef __attribute__((ext_vector_type(8)))  float    v8f;

#define BB   2
#define CC   576
#define HH   96
#define WW   96
#define NPIX (BB*HH*WW)      // 18432
#define C3   (3*CC)          // 1728
#define C2   (2*CC)          // 1152
#define C4   (CC/4)          // 144

__device__ __forceinline__ float wsum(float v){
#pragma unroll
  for(int m=16;m>0;m>>=1) v += __shfl_xor(v,m,32);
  return v;
}
__device__ __forceinline__ float wmax(float v){
#pragma unroll
  for(int m=16;m>0;m>>=1) v = fmaxf(v,__shfl_xor(v,m,32));
  return v;
}
__device__ __forceinline__ v16h cat8(v8h a, v8h b){
  v16h r;
#pragma unroll
  for(int i=0;i<8;i++){ r[i]=a[i]; r[i+8]=b[i]; }
  return r;
}
__device__ __forceinline__ v16h zero16(){
  v16h r;
#pragma unroll
  for(int i=0;i<16;i++) r[i]=(half_t)0.f;
  return r;
}
__device__ __forceinline__ float geluf(float x){
  return 0.5f*x*(1.f+erff(x*0.70710678118654752f));
}

// ---------------- layout / small kernels ----------------

__global__ void k_to_nhwc(const float* __restrict__ in, float* __restrict__ out){
  long idx = (long)blockIdx.x*blockDim.x + threadIdx.x;
  if (idx >= (long)NPIX*CC) return;
  int c = idx % CC; long p = idx / CC;
  int b = p / (HH*WW); int r = p % (HH*WW);
  out[idx] = in[((long)b*CC + c)*(HH*WW) + r];
}
__global__ void k_to_nchw(const float* __restrict__ in, float* __restrict__ out){
  long idx = (long)blockIdx.x*blockDim.x + threadIdx.x;
  if (idx >= (long)NPIX*CC) return;
  int w = idx % WW; int h = (idx/WW) % HH; int c = (idx/(WW*HH)) % CC; int b = idx/((long)WW*HH*CC);
  out[idx] = in[(((long)b*HH + h)*WW + w)*CC + c];
}

__global__ void k_zm(const float* __restrict__ z0, float* __restrict__ zm){
  int idx = blockIdx.x*blockDim.x + threadIdx.x;
  if (idx >= BB*HH*WW) return;
  int x = idx % WW; int y = (idx/WW) % HH; int b = idx/(HH*WW);
  float fy = y*0.5f - 0.25f, fx = x*0.5f - 0.25f;
  int y0 = (int)floorf(fy); float wy = fy - y0;
  int x0 = (int)floorf(fx); float wx = fx - x0;
  auto T = [&](int yy,int xx)->float{
    yy = min(max(yy,0),47); xx = min(max(xx,0),47);
    return z0[((long)b*48+yy)*48+xx] > 0.2f ? 1.f : 0.f;
  };
  zm[idx] = (1.f-wy)*((1.f-wx)*T(y0,x0)+wx*T(y0,x0+1))
          +      wy *((1.f-wx)*T(y0+1,x0)+wx*T(y0+1,x0+1));
}

__global__ void k_cast_h(const float* __restrict__ in, half_t* __restrict__ out, long n){
  long idx = (long)blockIdx.x*blockDim.x + threadIdx.x;
  if (idx < n) out[idx] = (half_t)in[idx];
}
// [O][C][k][k] -> [k*k][O][C] f16   (O=144, C=576)
__global__ void k_ckw(const float* __restrict__ in, half_t* __restrict__ out, int k){
  long total = (long)C4*CC*k*k;
  long idx = (long)blockIdx.x*blockDim.x + threadIdx.x;
  if (idx >= total) return;
  int kk = idx % (k*k); long rest = idx/(k*k);
  int c = rest % CC; int o = rest / CC;
  out[(long)kk*C4*CC + (long)o*CC + c] = (half_t)in[idx];
}

// per-pixel channel LayerNorm -> f16 NHWC
__global__ __launch_bounds__(256) void k_ln(const float* __restrict__ in,
    const float* __restrict__ g, const float* __restrict__ bsh, half_t* __restrict__ out){
  int wid = (blockIdx.x*blockDim.x + threadIdx.x) >> 5;
  int lane = threadIdx.x & 31;
  if (wid >= NPIX) return;
  const float* xp = in + (long)wid*CC;
  float s=0.f, s2=0.f;
#pragma unroll
  for (int c=lane; c<CC; c+=32){ float v=xp[c]; s+=v; s2+=v*v; }
  s = wsum(s); s2 = wsum(s2);
  float mu = s*(1.f/CC);
  float var = s2*(1.f/CC) - mu*mu;
  float rs = rsqrtf(var + 1e-5f);
  half_t* op = out + (long)wid*CC;
#pragma unroll
  for (int c=lane; c<CC; c+=32) op[c] = (half_t)((xp[c]-mu)*rs*g[c] + bsh[c]);
}

// depthwise 3x3 NHWC
__global__ void k_dw3(const float* __restrict__ in, const float* __restrict__ wgt,
                      const float* __restrict__ bias, float* __restrict__ out, int NCH){
  long idx = (long)blockIdx.x*blockDim.x + threadIdx.x;
  if (idx >= (long)NPIX*NCH) return;
  int c = idx % NCH; long p = idx / NCH;
  int b = p/(HH*WW); int r = p%(HH*WW); int y = r/WW; int x = r%WW;
  float s = bias[c];
#pragma unroll
  for (int ky=0;ky<3;ky++){
    int yy=y+ky-1; if (yy<0||yy>=HH) continue;
#pragma unroll
    for (int kx=0;kx<3;kx++){
      int xx=x+kx-1; if (xx<0||xx>=WW) continue;
      s += in[(((long)b*HH+yy)*WW+xx)*NCH + c] * wgt[(long)c*9 + ky*3+kx];
    }
  }
  out[idx] = s;
}

// attention-1 pooling: q1[b,c,w]=mean_h q ; k1[b,c,w]=max_h k
__global__ void k_pool1(const float* __restrict__ qkv2, float* __restrict__ q1, float* __restrict__ k1){
  int idx = blockIdx.x*blockDim.x + threadIdx.x;
  if (idx >= BB*CC*WW) return;
  int c = idx % CC; int w = (idx/CC) % WW; int b = idx/(CC*WW);
  float s=0.f, mx=-3.402823e38f;
  for (int h=0;h<HH;h++){
    long base = (((long)b*HH+h)*WW+w)*C3;
    s  += qkv2[base + c];
    mx  = fmaxf(mx, qkv2[base + CC + c]);
  }
  q1[((long)b*CC + c)*WW + w] = s*(1.f/HH);
  k1[((long)b*CC + c)*WW + w] = mx;
}

// attention-2 pooling: q2[b,h,w]=mean_c q ; k2 = mean_c k  (one wave per pixel)
__global__ __launch_bounds__(256) void k_pool2(const float* __restrict__ qkv2,
    float* __restrict__ q2, float* __restrict__ k2){
  int wid = (blockIdx.x*blockDim.x + threadIdx.x) >> 5;
  int lane = threadIdx.x & 31;
  if (wid >= NPIX) return;
  const float* p = qkv2 + (long)wid*C3;
  float sq=0.f, sk=0.f;
#pragma unroll
  for (int c=lane;c<CC;c+=32){ sq += p[c]; sk += p[CC+c]; }
  sq = wsum(sq); sk = wsum(sk);
  if (lane==0){ q2[wid]=sq*(1.f/CC); k2[wid]=sk*(1.f/CC); }
}

// l2 normalize rows of length 96 in place (one wave per row)
__global__ __launch_bounds__(256) void k_l2n(float* __restrict__ x, int rows){
  int wid = (blockIdx.x*blockDim.x + threadIdx.x) >> 5;
  int lane = threadIdx.x & 31;
  if (wid >= rows) return;
  float* r = x + (long)wid*WW;
  float v0=r[lane], v1=r[lane+32], v2=r[lane+64];
  float ss = wsum(v0*v0+v1*v1+v2*v2);
  float inv = 1.f / fmaxf(sqrtf(ss), 1e-12f);
  r[lane]=v0*inv; r[lane+32]=v1*inv; r[lane+64]=v2*inv;
}

__global__ void k_qz1(const float* __restrict__ zm, float* __restrict__ qz1){
  int idx = blockIdx.x*blockDim.x + threadIdx.x;
  if (idx >= BB*WW) return;
  int b = idx/WW, w = idx%WW;
  float s=0.f;
  for (int h=0;h<HH;h++) s += zm[((long)b*HH+h)*WW+w];
  qz1[idx] = s*(1.f/HH);
}
__global__ void k_az1(const float* __restrict__ qz1, const float* __restrict__ t, float* __restrict__ az){
  int idx = blockIdx.x*blockDim.x + threadIdx.x;
  if (idx >= BB*WW*WW) return;
  int w2 = idx%WW; int w1 = (idx/WW)%WW; int b = idx/(WW*WW);
  az[idx] = t[0]*qz1[b*WW+w1]*qz1[b*WW+w2];
}
// a1[b,w1,w2] = t * sum_c q1[b,c,w1]*k1[b,c,w2]
__global__ void k_aa1(const float* __restrict__ q1, const float* __restrict__ k1,
                      const float* __restrict__ t, float* __restrict__ a1){
  int idx = blockIdx.x*blockDim.x + threadIdx.x;
  if (idx >= BB*WW*WW) return;
  int w2 = idx%WW; int w1 = (idx/WW)%WW; int b = idx/(WW*WW);
  float s=0.f;
  for (int c=0;c<CC;c++)
    s += q1[((long)b*CC+c)*WW+w1]*k1[((long)b*CC+c)*WW+w2];
  a1[idx] = t[0]*s;
}
// out[b,i,j] = t * sum_w A[b,i,w]*Bp[b,j,w]   (rows of length 96)
__global__ void k_dotrow(const float* __restrict__ A, const float* __restrict__ Bp,
                         const float* __restrict__ t, float* __restrict__ out){
  int idx = blockIdx.x*blockDim.x + threadIdx.x;
  if (idx >= BB*HH*HH) return;
  int j = idx%HH; int i = (idx/HH)%HH; int b = idx/(HH*HH);
  float s=0.f;
  for (int w=0;w<WW;w++)
    s += A[((long)b*HH+i)*WW+w]*Bp[((long)b*HH+j)*WW+w];
  out[idx] = t[0]*s;
}

// softmax(A)+softmax(Bm) row-wise (rows len 96), store TRANSPOSED f16: outT[b][col][row]
__global__ __launch_bounds__(256) void k_smaxT(const float* __restrict__ A,
    const float* __restrict__ Bm, half_t* __restrict__ outT){
  int wid = (blockIdx.x*blockDim.x + threadIdx.x) >> 5;
  int lane = threadIdx.x & 31;
  if (wid >= BB*96) return;
  int b = wid/96, r = wid%96;
  const float* ar = A  + ((long)b*96 + r)*96;
  const float* br = Bm + ((long)b*96 + r)*96;
  float a0=ar[lane], a1=ar[lane+32], a2=ar[lane+64];
  float b0=br[lane], b1=br[lane+32], b2=br[lane+64];
  float ma = wmax(fmaxf(a0,fmaxf(a1,a2)));
  float mb = wmax(fmaxf(b0,fmaxf(b1,b2)));
  float e0=expf(a0-ma), e1=expf(a1-ma), e2=expf(a2-ma);
  float f0=expf(b0-mb), f1=expf(b1-mb), f2=expf(b2-mb);
  float sa = wsum(e0+e1+e2), sb = wsum(f0+f1+f2);
  float ia = 1.f/sa, ib = 1.f/sb;
  half_t* op = outT + (long)b*96*96;
  op[(lane   )*96 + r] = (half_t)(e0*ia + f0*ib);
  op[(lane+32)*96 + r] = (half_t)(e1*ia + f1*ib);
  op[(lane+64)*96 + r] = (half_t)(e2*ia + f2*ib);
}

// v slice (channels 2C..3C of qkv2 NHWC) -> f16 [b][h][c][w]
__global__ void k_vtr1(const float* __restrict__ qkv2, half_t* __restrict__ vt){
  long idx = (long)blockIdx.x*blockDim.x + threadIdx.x;
  if (idx >= (long)BB*HH*CC*WW) return;
  int w = idx%WW; int c = (idx/WW)%CC; int h = (idx/((long)WW*CC))%HH; int b = idx/((long)WW*CC*HH);
  vt[idx] = (half_t)qkv2[(((long)b*HH+h)*WW+w)*C3 + 2*CC + c];
}
// v slice -> f16 [b][w][c][h]
__global__ void k_vtr2(const float* __restrict__ qkv2, half_t* __restrict__ vt){
  long idx = (long)blockIdx.x*blockDim.x + threadIdx.x;
  if (idx >= (long)BB*WW*CC*HH) return;
  int h = idx%HH; int c = (idx/HH)%CC; int w = (idx/((long)HH*CC))%WW; int b = idx/((long)HH*CC*WW);
  vt[idx] = (half_t)qkv2[(((long)b*HH+h)*WW+w)*C3 + 2*CC + c];
}
// x2 slice (channels C..2C of ffn dwconv out, stride 2C) -> f16 [pix][C]
__global__ void k_x2h(const float* __restrict__ in, half_t* __restrict__ out){
  long idx = (long)blockIdx.x*blockDim.x + threadIdx.x;
  if (idx >= (long)NPIX*CC) return;
  int c = idx%CC; long p = idx/CC;
  out[idx] = (half_t)in[p*C2 + CC + c];
}

// ---------------- WMMA GEMM kernels ----------------
// D[n,o] = sum_{kk,c} Wf[kk][o][c] * Xf[shift(n,kk)][c]  (+bias, epilogue EPI)
// EPI: 0 plain f32  1 f32 + residual  2 f16 store  3 FFN: z + gelu(x1)*sigmoid(v)
template<int KS, int EPI>
__global__ __launch_bounds__(256) void gemm_conv(
    const half_t* __restrict__ Wf, const half_t* __restrict__ Xf,
    const float* __restrict__ bias,
    const float* __restrict__ res, int res_stride,
    const float* __restrict__ res2,
    float* __restrict__ outF, half_t* __restrict__ outH,
    int O, int Cin, int ostride, int ooffset)
{
  const int wave = (blockIdx.x*blockDim.x + threadIdx.x) >> 5;
  const int lane = threadIdx.x & 31;
  const int mtiles = O >> 4;
  const int mt = wave % mtiles;
  const int ng = wave / mtiles;
  const int m0 = mt << 4;
  const int n0 = ng << 6;
  const int lm = lane & 15, hi = lane >> 4;

  v8f acc[4];
#pragma unroll
  for (int t=0;t<4;t++)
#pragma unroll
    for (int r=0;r<8;r++) acc[t][r]=0.f;

  int pb[4], py[4], px[4];
#pragma unroll
  for (int t=0;t<4;t++){
    int n = n0 + t*16 + lm;
    pb[t] = n/(HH*WW); int rp = n%(HH*WW); py[t]=rp/WW; px[t]=rp%WW;
  }

  for (int kk=0; kk<KS*KS; ++kk){
    const int dy = kk/KS - KS/2, dx = kk%KS - KS/2;
    const half_t* Wk = Wf + ((long)kk*O + (m0+lm))*Cin;
    for (int c0=0; c0<Cin; c0+=32){
      const half_t* ap = Wk + c0 + hi*16;
      v16h a = cat8(*(const v8h*)ap, *(const v8h*)(ap+8));
#pragma unroll
      for (int t=0;t<4;t++){
        v16h bf;
        int yy = py[t]+dy, xx = px[t]+dx;
        bool ok = (KS==1) || (yy>=0 && yy<HH && xx>=0 && xx<WW);
        if (ok){
          long ps = ((long)pb[t]*HH + yy)*WW + xx;
          const half_t* bp = Xf + ps*Cin + c0 + hi*16;
          bf = cat8(*(const v8h*)bp, *(const v8h*)(bp+8));
        } else {
          bf = zero16();
        }
        acc[t] = __builtin_amdgcn_wmma_f32_16x16x32_f16(false, a, false, bf,
                                                        (short)0, acc[t], false, false);
      }
    }
  }

#pragma unroll
  for (int t=0;t<4;t++){
    const long n = n0 + t*16 + lm;
    const int mb = m0 + hi*8;
    const long ob = n*(long)ostride + ooffset + mb;
#pragma unroll
    for (int r=0;r<8;r++){
      float v = acc[t][r] + bias[mb+r];
      if (EPI==0){
        outF[ob+r] = v;
      } else if (EPI==1){
        outF[ob+r] = v + res[n*(long)res_stride + mb + r];
      } else if (EPI==2){
        outH[ob+r] = (half_t)v;
      } else { // EPI==3
        float sg = 1.f/(1.f+expf(-v));
        float x1v = res[n*(long)res_stride + mb + r];
        float zv  = res2[n*(long)ostride + mb + r];
        outF[ob+r] = zv + geluf(x1v)*sg;
      }
    }
  }
}

// batched per-b: O[m,n] = sum_k Vt[b][m][k] * AttnT[b][n][k]; M=W*C per b, N=K=96
// store f16 at  b*H*W*C + (m/C)*s_m + n*s_n + (m%C)
__global__ __launch_bounds__(256) void gemm_attn(
    const half_t* __restrict__ Vt, const half_t* __restrict__ AT,
    half_t* __restrict__ outH, long s_m, long s_n)
{
  const int wave = (blockIdx.x*blockDim.x + threadIdx.x) >> 5;
  const int lane = threadIdx.x & 31;
  const int MT = (WW*CC)>>4;   // 3456 m-tiles per b
  const int b  = wave / MT;
  const int m0 = (wave % MT) << 4;
  const int lm = lane & 15, hi = lane >> 4;

  v8f acc[6];
#pragma unroll
  for (int t=0;t<6;t++)
#pragma unroll
    for (int r=0;r<8;r++) acc[t][r]=0.f;

  const half_t* Vb = Vt + (long)b*WW*CC*96;
  const half_t* Ab = AT + (long)b*96*96;
#pragma unroll
  for (int k0=0;k0<96;k0+=32){
    const half_t* ap = Vb + (long)(m0+lm)*96 + k0 + hi*16;
    v16h a = cat8(*(const v8h*)ap, *(const v8h*)(ap+8));
#pragma unroll
    for (int nt=0;nt<6;nt++){
      const half_t* bp = Ab + (long)(nt*16+lm)*96 + k0 + hi*16;
      v16h bf = cat8(*(const v8h*)bp, *(const v8h*)(bp+8));
      acc[nt] = __builtin_amdgcn_wmma_f32_16x16x32_f16(false, a, false, bf,
                                                       (short)0, acc[nt], false, false);
    }
  }

  const long obase = (long)b*HH*WW*CC;
#pragma unroll
  for (int nt=0;nt<6;nt++){
    const int n = nt*16 + lm;
#pragma unroll
    for (int r=0;r<8;r++){
      int m = m0 + hi*8 + r;
      long off = obase + (long)(m/CC)*s_m + (long)n*s_n + (m%CC);
      outH[off] = (half_t)acc[nt][r];
    }
  }
}

// ---------------- driver ----------------

static inline int cdiv(long a, long b){ return (int)((a+b-1)/b); }

extern "C" void kernel_launch(void* const* d_in, const int* in_sizes, int n_in,
                              void* d_out, int out_size, void* d_ws, size_t ws_size,
                              hipStream_t stream) {
  (void)in_sizes; (void)n_in; (void)out_size; (void)ws_size;
  const float* x_in  = (const float*)d_in[0];
  const float* z0_in = (const float*)d_in[1];

  char* ws = (char*)d_ws; size_t off = 0;
  auto alloc = [&](size_t bytes)->void*{
    off = (off + 255) & ~(size_t)255;
    void* p = ws + off; off += bytes; return p;
  };

  float*  zm    = (float*) alloc((long)BB*HH*WW*4);
  float*  X0    = (float*) alloc((long)NPIX*CC*4);
  float*  X1    = (float*) alloc((long)NPIX*CC*4);
  half_t* lnH   = (half_t*)alloc((long)NPIX*CC*2);      // also oH / x2 f16
  float*  qkvb  = (float*) alloc((long)NPIX*C3*4);      // also pi out
  float*  qkv2  = (float*) alloc((long)NPIX*C3*4);      // dwconv out / ffn dwconv out
  half_t* vt    = (half_t*)alloc((long)BB*CC*HH*WW*2);  // v transposed / g buffer
  half_t* attnT = (half_t*)alloc((long)BB*96*96*2);
  float*  q1    = (float*) alloc((long)BB*CC*WW*4);
  float*  k1    = (float*) alloc((long)BB*CC*WW*4);
  float*  q2    = (float*) alloc((long)BB*HH*WW*4);
  float*  k2    = (float*) alloc((long)BB*HH*WW*4);
  float*  qz1   = (float*) alloc((long)BB*WW*4);
  float*  a1b   = (float*) alloc((long)BB*96*96*4);
  float*  azb   = (float*) alloc((long)BB*96*96*4);
  half_t* qkvW  = (half_t*)alloc((long)C3*CC*2);
  half_t* poW   = (half_t*)alloc((long)CC*CC*2);
  half_t* piW   = (half_t*)alloc((long)C2*CC*2);
  half_t* cfW   = (half_t*)alloc((long)CC*CC*2);
  half_t* ckW   = (half_t*)alloc((long)(1+9+25+49)*C4*CC*2);

  const int TPB = 256;
  auto GE = [&](long n){ return cdiv(n, TPB); };
  auto GW = [&](long waves){ return (int)(waves/8); }; // 8 waves per 256-thr block (exact)

  k_to_nhwc<<<GE((long)NPIX*CC),TPB,0,stream>>>(x_in, X0);
  k_zm     <<<GE((long)BB*HH*WW),TPB,0,stream>>>(z0_in, zm);

  float* cur = X0; float* alt = X1;
  const int NLAYERS = 2;

  for (int L=0; L<NLAYERS; ++L){
    const float* P[28];
    for (int i=0;i<28;i++) P[i] = (const float*)d_in[2 + 28*L + i];
    const float *c1_b=P[0], *c1_w=P[1], *c3_b=P[2], *c3_w=P[3], *c5_b=P[4], *c5_w=P[5],
                *c7_b=P[6], *c7_w=P[7], *cf_b=P[8], *cf_w=P[9], *dw2_b=P[10], *dw2_w=P[11],
                *dw_b=P[12], *dw_w=P[13], *ln1_b=P[14], *ln1_w=P[15], *ln2_b=P[16], *ln2_w=P[17],
                *pi_b=P[18], *pi_w=P[19], *po_b=P[20], *po_w=P[21], *qkv_b=P[22], *qkv_w=P[23],
                *t1=P[24], *t2=P[25], *t3=P[26], *t4=P[27];

    // weight conversions (fp32 -> f16, GEMM layouts)
    k_cast_h<<<GE((long)C3*CC),TPB,0,stream>>>(qkv_w, qkvW, (long)C3*CC);
    k_cast_h<<<GE((long)CC*CC),TPB,0,stream>>>(po_w,  poW,  (long)CC*CC);
    k_cast_h<<<GE((long)C2*CC),TPB,0,stream>>>(pi_w,  piW,  (long)C2*CC);
    k_cast_h<<<GE((long)CC*CC),TPB,0,stream>>>(cf_w,  cfW,  (long)CC*CC);
    half_t* ck1 = ckW;
    half_t* ck3 = ck1 + (long)1 *C4*CC;
    half_t* ck5 = ck3 + (long)9 *C4*CC;
    half_t* ck7 = ck5 + (long)25*C4*CC;
    k_ckw<<<GE((long)C4*CC*1 ),TPB,0,stream>>>(c1_w, ck1, 1);
    k_ckw<<<GE((long)C4*CC*9 ),TPB,0,stream>>>(c3_w, ck3, 3);
    k_ckw<<<GE((long)C4*CC*25),TPB,0,stream>>>(c5_w, ck5, 5);
    k_ckw<<<GE((long)C4*CC*49),TPB,0,stream>>>(c7_w, ck7, 7);

    const long NG = NPIX/64; // 288 pixel groups

    // ---------- attention 1 ----------
    k_ln<<<GE((long)NPIX*32),TPB,0,stream>>>(cur, ln1_w, ln1_b, lnH);
    gemm_conv<1,0><<<GW((C3/16)*NG),TPB,0,stream>>>(qkvW, lnH, qkv_b,
        nullptr,0,nullptr, qkvb, nullptr, C3, CC, C3, 0);
    k_dw3<<<GE((long)NPIX*C3),TPB,0,stream>>>(qkvb, dw_w, dw_b, qkv2, C3);

    k_pool1<<<GE((long)BB*CC*WW),TPB,0,stream>>>(qkv2, q1, k1);
    k_l2n<<<GE((long)BB*CC*32),TPB,0,stream>>>(q1, BB*CC);
    k_l2n<<<GE((long)BB*CC*32),TPB,0,stream>>>(k1, BB*CC);
    k_qz1<<<GE((long)BB*WW),TPB,0,stream>>>(zm, qz1);
    k_aa1<<<GE((long)BB*96*96),TPB,0,stream>>>(q1, k1, t1, a1b);
    k_az1<<<GE((long)BB*96*96),TPB,0,stream>>>(qz1, t2, azb);
    k_smaxT<<<GE((long)BB*96*32),TPB,0,stream>>>(a1b, azb, attnT);

    k_vtr1<<<GE((long)BB*HH*CC*WW),TPB,0,stream>>>(qkv2, vt);
    gemm_attn<<<GW((long)BB*((WW*CC)/16)),TPB,0,stream>>>(vt, attnT, lnH,
        (long)WW*CC /*s_m over h*/, (long)CC /*s_n over w2*/);
    gemm_conv<1,1><<<GW((CC/16)*NG),TPB,0,stream>>>(poW, lnH, po_b,
        cur, CC, nullptr, alt, nullptr, CC, CC, CC, 0);   // out1 = x + po(o)

    // ---------- attention 2 ----------
    k_ln<<<GE((long)NPIX*32),TPB,0,stream>>>(alt, ln1_w, ln1_b, lnH);
    gemm_conv<1,0><<<GW((C3/16)*NG),TPB,0,stream>>>(qkvW, lnH, qkv_b,
        nullptr,0,nullptr, qkvb, nullptr, C3, CC, C3, 0);
    k_dw3<<<GE((long)NPIX*C3),TPB,0,stream>>>(qkvb, dw_w, dw_b, qkv2, C3);

    k_pool2<<<GE((long)NPIX*32),TPB,0,stream>>>(qkv2, q2, k2);
    k_l2n<<<GE((long)BB*HH*32),TPB,0,stream>>>(q2, BB*HH);
    k_l2n<<<GE((long)BB*HH*32),TPB,0,stream>>>(k2, BB*HH);
    k_dotrow<<<GE((long)BB*96*96),TPB,0,stream>>>(q2, k2, t3, a1b);
    k_dotrow<<<GE((long)BB*96*96),TPB,0,stream>>>(zm, zm, t4, azb);
    k_smaxT<<<GE((long)BB*96*32),TPB,0,stream>>>(a1b, azb, attnT);

    k_vtr2<<<GE((long)BB*WW*CC*HH),TPB,0,stream>>>(qkv2, vt);
    gemm_attn<<<GW((long)BB*((WW*CC)/16)),TPB,0,stream>>>(vt, attnT, lnH,
        (long)CC /*s_m over w*/, (long)WW*CC /*s_n over h2*/);
    gemm_conv<1,1><<<GW((CC/16)*NG),TPB,0,stream>>>(poW, lnH, po_b,
        alt, CC, nullptr, cur, nullptr, CC, CC, CC, 0);   // z = out1 + po(o)

    // ---------- FFN ----------
    k_ln<<<GE((long)NPIX*32),TPB,0,stream>>>(cur, ln2_w, ln2_b, lnH);
    gemm_conv<1,0><<<GW((C2/16)*NG),TPB,0,stream>>>(piW, lnH, pi_b,
        nullptr,0,nullptr, qkvb, nullptr, C2, CC, C2, 0);
    k_dw3<<<GE((long)NPIX*C2),TPB,0,stream>>>(qkvb, dw2_w, dw2_b, qkv2, C2);
    k_x2h<<<GE((long)NPIX*CC),TPB,0,stream>>>(qkv2, lnH);  // x2 -> f16

    // multiscale convs -> g (f16 in vt), channel offsets 0/144/288/432
    gemm_conv<1,2><<<GW((C4/16)*NG),TPB,0,stream>>>(ck1, lnH, c1_b,
        nullptr,0,nullptr, nullptr, vt, C4, CC, CC, 0*C4);
    gemm_conv<3,2><<<GW((C4/16)*NG),TPB,0,stream>>>(ck3, lnH, c3_b,
        nullptr,0,nullptr, nullptr, vt, C4, CC, CC, 1*C4);
    gemm_conv<5,2><<<GW((C4/16)*NG),TPB,0,stream>>>(ck5, lnH, c5_b,
        nullptr,0,nullptr, nullptr, vt, C4, CC, CC, 2*C4);
    gemm_conv<7,2><<<GW((C4/16)*NG),TPB,0,stream>>>(ck7, lnH, c7_b,
        nullptr,0,nullptr, nullptr, vt, C4, CC, CC, 3*C4);

    // out = z + gelu(x1) * sigmoid(cf(g))
    gemm_conv<1,3><<<GW((CC/16)*NG),TPB,0,stream>>>(cfW, vt, cf_b,
        qkv2 /*x1*/, C2, cur /*z*/, alt, nullptr, CC, CC, CC, 0);

    float* tmp = cur; cur = alt; alt = tmp;
  }

  k_to_nchw<<<GE((long)NPIX*CC),TPB,0,stream>>>(cur, (float*)d_out);
}
